// Net_1047972020276
// MI455X (gfx1250) — matrix operand order repaired and checked by reading, not compiled, not run on hardware
//
#include <hip/hip_runtime.h>
#include <hip/hip_bf16.h>
#include <math.h>

typedef __attribute__((ext_vector_type(16))) _Float16     v16h;
typedef __attribute__((ext_vector_type(8)))  float        v8f;
typedef __attribute__((ext_vector_type(8)))  unsigned int v8u;

#define N_NODES 100000
#define F_IN    100

// ---------------------------------------------------------------------------
// Zero-fill (graph-capture-safe alternative to memset)
// ---------------------------------------------------------------------------
__global__ void zero_f32(float* __restrict__ p, long long n) {
  long long i = (long long)blockIdx.x * blockDim.x + threadIdx.x;
  if (i < n) p[i] = 0.0f;
}

// ---------------------------------------------------------------------------
// Degree (incoming edges at dst) and dinv = rsqrt(deg + 1)
// ---------------------------------------------------------------------------
__global__ void deg_kernel(const int* __restrict__ dst, float* __restrict__ deg,
                           long long E) {
  long long e = (long long)blockIdx.x * blockDim.x + threadIdx.x;
  if (e < E) atomicAdd(&deg[dst[e]], 1.0f);
}

__global__ void dinv_kernel(float* __restrict__ deg, int n) {
  int i = blockIdx.x * blockDim.x + threadIdx.x;
  if (i < n) deg[i] = rsqrtf(deg[i] + 1.0f);
}

// ---------------------------------------------------------------------------
// Layer-1 GEMM via WMMA: H[N,32] = X[N,100] * W[100,32]  (f16 math, f32 acc)
// Block = 128 threads (4 waves). Block computes a 64x32 output tile.
// X tile staged in LDS as 64x128 f16 (zero-padded K: 100 -> 128);
// W staged TRANSPOSED as 32x128 f16 so B K-pairs are contiguous in LDS.
// Each wave: one 16-row tile x both 16-col tiles -> 8 chained WMMAs.
// Full-tile blocks take an unguarded fast path (block-uniform branch).
// ---------------------------------------------------------------------------
__global__ __launch_bounds__(128)
void gemm_wmma_l1(const float* __restrict__ X, const float* __restrict__ W,
                  float* __restrict__ H, int N) {
  __shared__ _Float16 Xs[64 * 128];   // [row][k]   16 KB
  __shared__ _Float16 WsT[32 * 128];  // [n][k]      8 KB

  const int  tid  = threadIdx.x;       // 0..127
  const int  m0   = blockIdx.x * 64;   // block row base
  const bool full = (m0 + 64 <= N);    // block-uniform: no row guards needed

  // ---- Stage X tile: thread t owns column k = t for all 64 rows ----
  {
    const bool kvalid = (tid < F_IN);
    if (full) {
      #pragma unroll 4
      for (int row = 0; row < 64; ++row) {
        float v = kvalid ? X[(size_t)(m0 + row) * F_IN + tid] : 0.0f;
        Xs[row * 128 + tid] = (_Float16)v;
      }
    } else {
      #pragma unroll 4
      for (int row = 0; row < 64; ++row) {
        int grow = m0 + row;
        float v = 0.0f;
        if (kvalid && grow < N) v = X[(size_t)grow * F_IN + tid];
        Xs[row * 128 + tid] = (_Float16)v;
      }
    }
  }
  // ---- Stage W transposed: thread t owns k = t for all 32 columns ----
  {
    const bool kvalid = (tid < F_IN);
    #pragma unroll 4
    for (int n = 0; n < 32; ++n) {
      float v = kvalid ? W[(size_t)tid * 32 + n] : 0.0f;
      WsT[n * 128 + tid] = (_Float16)v;
    }
  }
  __syncthreads();

  const int wv    = tid >> 5;              // wave id 0..3 -> row sub-tile
  const int lane  = tid & 31;
  const int lrow  = wv * 16 + (lane & 15); // A: local row in Xs
  const int khalf = lane >> 4;             // K-half select
  const int ncol  = lane & 15;             // B/D: column within 16-wide tile

  v8f acc0 = {}, acc1 = {};

  #pragma unroll
  for (int k0 = 0; k0 < 128; k0 += 32) {
    v8u au, bu0, bu1;
    // A fragment: VGPR v holds K-pair kb,kb+1 (contiguous f16 -> b32/b128 LDS reads)
    #pragma unroll
    for (int v = 0; v < 8; ++v) {
      int kb = k0 + ((v < 4) ? 0 : 16) + khalf * 8 + (v & 3) * 2;
      au[v] = *reinterpret_cast<const unsigned int*>(&Xs[lrow * 128 + kb]);
    }
    // B fragments (two 16-col tiles): same K-pair order as A
    #pragma unroll
    for (int v = 0; v < 8; ++v) {
      int kb = k0 + ((v < 4) ? 0 : 16) + khalf * 8 + (v & 3) * 2;
      bu0[v] = *reinterpret_cast<const unsigned int*>(&WsT[ncol * 128 + kb]);
      bu1[v] = *reinterpret_cast<const unsigned int*>(&WsT[(16 + ncol) * 128 + kb]);
    }
    v16h a = __builtin_bit_cast(v16h, au);
    acc0 = __builtin_amdgcn_wmma_f32_16x16x32_f16(
        false, a, false, __builtin_bit_cast(v16h, bu0), (short)0, acc0, false, false);
    acc1 = __builtin_amdgcn_wmma_f32_16x16x32_f16(
        false, a, false, __builtin_bit_cast(v16h, bu1), (short)0, acc1, false, false);
  }

  // D layout: VGPR r -> row = rbase + r, col = ncol (acc0) / ncol+16 (acc1)
  const int rbase = m0 + wv * 16 + khalf * 8;
  if (full) {
    float* __restrict__ Hp = H + (size_t)rbase * 32 + ncol;
    #pragma unroll
    for (int r = 0; r < 8; ++r) {
      Hp[r * 32]      = acc0[r];
      Hp[r * 32 + 16] = acc1[r];
    }
  } else {
    #pragma unroll
    for (int r = 0; r < 8; ++r) {
      int row = rbase + r;
      if (row < N) {
        H[(size_t)row * 32 + ncol]      = acc0[r];
        H[(size_t)row * 32 + 16 + ncol] = acc1[r];
      }
    }
  }
}

// ---------------------------------------------------------------------------
// Small dense linear layers (K <= 32, Fout <= 4): per-node VALU FMAs.
// ---------------------------------------------------------------------------
template <int FinT, int FoutT>
__global__ void lin_small(const float* __restrict__ Xin, const float* __restrict__ W,
                          float* __restrict__ H, int N) {
  int i = blockIdx.x * blockDim.x + threadIdx.x;
  if (i >= N) return;
  float xin[FinT];
  #pragma unroll
  for (int k = 0; k < FinT; ++k) xin[k] = Xin[(size_t)i * FinT + k];
  #pragma unroll
  for (int f = 0; f < FoutT; ++f) {
    float acc = 0.0f;
    #pragma unroll
    for (int k = 0; k < FinT; ++k) acc = fmaf(xin[k], W[k * FoutT + f], acc);
    H[(size_t)i * FoutT + f] = acc;
  }
}

// ---------------------------------------------------------------------------
// Edge aggregation: agg[dst] += h[src] * dinv[src]*dinv[dst]
// Feature buffers fit in 192MB L2 -> gathers + float atomics stay on-chip.
// ---------------------------------------------------------------------------
template <int F>
__global__ void edge_agg(const int* __restrict__ src, const int* __restrict__ dst,
                         const float* __restrict__ dinv, const float* __restrict__ h,
                         float* __restrict__ agg, long long E) {
  long long e = (long long)blockIdx.x * blockDim.x + threadIdx.x;
  if (e >= E) return;
  int s = src[e], d = dst[e];
  float norm = dinv[s] * dinv[d];
  const float* hs = h + (size_t)s * F;
  float* ad = agg + (size_t)d * F;
  #pragma unroll
  for (int f = 0; f < F; ++f) atomicAdd(&ad[f], hs[f] * norm);
}

// ---------------------------------------------------------------------------
// Node finalize: out = tanh(agg + hlin*dinv^2 + b)
// ---------------------------------------------------------------------------
template <int F>
__global__ void node_tanh(const float* __restrict__ agg, const float* __restrict__ hlin,
                          const float* __restrict__ dinv, const float* __restrict__ b,
                          float* __restrict__ out, int N) {
  int i = blockIdx.x * blockDim.x + threadIdx.x;
  if (i >= N) return;
  float di2 = dinv[i] * dinv[i];
  #pragma unroll
  for (int f = 0; f < F; ++f) {
    float v = agg[(size_t)i * F + f] + hlin[(size_t)i * F + f] * di2 + b[f];
    out[(size_t)i * F + f] = tanhf(v);
  }
}

// Final layer: v = agg + hlin*dinv^2 + b, then log_softmax over 4 features
__global__ void node_logsoftmax4(const float* __restrict__ agg, const float* __restrict__ hlin,
                                 const float* __restrict__ dinv, const float* __restrict__ b,
                                 float* __restrict__ y, int N) {
  int i = blockIdx.x * blockDim.x + threadIdx.x;
  if (i >= N) return;
  float di2 = dinv[i] * dinv[i];
  float v[4];
  #pragma unroll
  for (int f = 0; f < 4; ++f)
    v[f] = agg[(size_t)i * 4 + f] + hlin[(size_t)i * 4 + f] * di2 + b[f];
  float m = fmaxf(fmaxf(v[0], v[1]), fmaxf(v[2], v[3]));
  float s = expf(v[0] - m) + expf(v[1] - m) + expf(v[2] - m) + expf(v[3] - m);
  float lse = m + logf(s);
  #pragma unroll
  for (int f = 0; f < 4; ++f) y[(size_t)i * 4 + f] = v[f] - lse;
}

// ---------------------------------------------------------------------------
// Launch
// ---------------------------------------------------------------------------
extern "C" void kernel_launch(void* const* d_in, const int* in_sizes, int n_in,
                              void* d_out, int out_size, void* d_ws, size_t ws_size,
                              hipStream_t stream) {
  const float* x  = (const float*)d_in[0];
  const int*   ei = (const int*)d_in[1];
  const float* W1 = (const float*)d_in[2];
  const float* b1 = (const float*)d_in[3];
  const float* W2 = (const float*)d_in[4];
  const float* b2 = (const float*)d_in[5];
  const float* W3 = (const float*)d_in[6];
  const float* b3 = (const float*)d_in[7];
  const float* W4 = (const float*)d_in[8];
  const float* b4 = (const float*)d_in[9];

  const int       N = N_NODES;
  const long long E = (long long)in_sizes[1] / 2;
  const int*   src  = ei;
  const int*   dst  = ei + E;

  float* ws   = (float*)d_ws;
  float* dinv = ws;                    // N        (degree, then dinv in place)
  float* bufA = dinv + N;              // N*32     (linear output of current layer)
  float* bufB = bufA + (size_t)N * 32; // N*32     (aggregation accumulator)
  float* bufC = bufB + (size_t)N * 32; // N*32     (activated output -> next input)

  float* h_out = (float*)d_out;            // [N,2]  (layer-3 activations)
  float* y_out = (float*)d_out + 2 * N;    // [N,4]  (log-softmax)

  const int TB = 256;
  auto blocksN = [&](long long n) { return (unsigned)((n + TB - 1) / TB); };
  const unsigned eb = blocksN(E);

  // --- degree + dinv ---
  zero_f32<<<blocksN(N), TB, 0, stream>>>(dinv, N);
  deg_kernel<<<eb, TB, 0, stream>>>(dst, dinv, E);
  dinv_kernel<<<blocksN(N), TB, 0, stream>>>(dinv, N);

  // --- Layer 1: 100 -> 32 (WMMA GEMM, LDS-staged) ---
  gemm_wmma_l1<<<(N + 63) / 64, 128, 0, stream>>>(x, W1, bufA, N);
  zero_f32<<<blocksN((long long)N * 32), TB, 0, stream>>>(bufB, (long long)N * 32);
  edge_agg<32><<<eb, TB, 0, stream>>>(src, dst, dinv, bufA, bufB, E);
  node_tanh<32><<<blocksN(N), TB, 0, stream>>>(bufB, bufA, dinv, b1, bufC, N);

  // --- Layer 2: 32 -> 4 ---
  lin_small<32, 4><<<blocksN(N), TB, 0, stream>>>(bufC, W2, bufA, N);
  zero_f32<<<blocksN((long long)N * 4), TB, 0, stream>>>(bufB, (long long)N * 4);
  edge_agg<4><<<eb, TB, 0, stream>>>(src, dst, dinv, bufA, bufB, E);
  node_tanh<4><<<blocksN(N), TB, 0, stream>>>(bufB, bufA, dinv, b2, bufC, N);

  // --- Layer 3: 4 -> 2  (activated output goes straight to d_out "h") ---
  lin_small<4, 2><<<blocksN(N), TB, 0, stream>>>(bufC, W3, bufA, N);
  zero_f32<<<blocksN((long long)N * 2), TB, 0, stream>>>(bufB, (long long)N * 2);
  edge_agg<2><<<eb, TB, 0, stream>>>(src, dst, dinv, bufA, bufB, E);
  node_tanh<2><<<blocksN(N), TB, 0, stream>>>(bufB, bufA, dinv, b3, h_out, N);

  // --- Layer 4: 2 -> 4, then log_softmax ---
  lin_small<2, 4><<<blocksN(N), TB, 0, stream>>>(h_out, W4, bufA, N);
  zero_f32<<<blocksN((long long)N * 4), TB, 0, stream>>>(bufB, (long long)N * 4);
  edge_agg<4><<<eb, TB, 0, stream>>>(src, dst, dinv, bufA, bufB, E);
  node_logsoftmax4<<<blocksN(N), TB, 0, stream>>>(bufB, bufA, dinv, b4, y_out, N);
}
